// Encoder_2757369004690
// MI455X (gfx1250) — matrix operand order repaired and checked by reading, not compiled, not run on hardware
//
#include <hip/hip_runtime.h>
#include <hip/hip_bf16.h>

// ---------------------------------------------------------------------------
// Problem constants (match reference)
// ---------------------------------------------------------------------------
constexpr int N    = 16384;
constexpr int K    = 25;
constexpr int R    = 3;
constexpr int D    = 256;
constexpr int OUT  = 256;
constexpr int NTOT = 100000;
constexpr int BAG  = 8;
constexpr int KC   = (R + 1) * D;   // 1024 = concat K dim of the fused GEMM

typedef __attribute__((ext_vector_type(16))) __bf16 v16bf;
typedef __attribute__((ext_vector_type(8)))  __bf16 v8bf;
typedef __attribute__((ext_vector_type(8)))  float  v8f;
typedef __attribute__((ext_vector_type(4)))  int    v4i;

// The async-copy builtin takes v4i pointers in AS1 (global) / AS3 (LDS).
typedef __attribute__((address_space(1))) v4i glb_v4i;
typedef __attribute__((address_space(3))) v4i lds_v4i;

// float -> bf16 with round-to-nearest-even (ignoring NaN edge case)
__device__ __forceinline__ unsigned short f2bf(float f) {
    union { float f; unsigned int u; } v; v.f = f;
    unsigned int r = v.u + 0x7FFFu + ((v.u >> 16) & 1u);
    return (unsigned short)(r >> 16);
}
__device__ __forceinline__ unsigned int pack2bf(float a, float b) {
    return (unsigned int)f2bf(a) | ((unsigned int)f2bf(b) << 16);
}

__device__ __forceinline__ v16bf cat8(v8bf lo, v8bf hi) {
    return __builtin_shufflevector(lo, hi,
        0, 1, 2, 3, 4, 5, 6, 7, 8, 9, 10, 11, 12, 13, 14, 15);
}

// 16B async global->LDS copy (ASYNCcnt path); VGPR-roundtrip fallback.
__device__ __forceinline__ void async_copy16(const __bf16* g, unsigned char* l) {
#if __has_builtin(__builtin_amdgcn_global_load_async_to_lds_b128)
    __builtin_amdgcn_global_load_async_to_lds_b128(
        (glb_v4i*)(uintptr_t)g, (lds_v4i*)l, 0, 0);
#else
    *(uint4*)l = *(const uint4*)g;
#endif
}
__device__ __forceinline__ void async_wait_all() {
#if __has_builtin(__builtin_amdgcn_s_wait_asynccnt)
    __builtin_amdgcn_s_wait_asynccnt(0);
#endif
}

// ---------------------------------------------------------------------------
// Kernel 1: gather + EmbeddingBag-mean, emit bf16 X = [mean0|mean1|mean2|self]
// Layout: X[n, s*256 + d].  One block per node; 4 sections x 64 float4 lanes.
// Each neighbor row is a coalesced 1KB burst; ~1.27 GB of gathers is the
// HBM-bound floor (~55 us @ 23.3 TB/s) for the whole problem.
// ---------------------------------------------------------------------------
__global__ void __launch_bounds__(256)
gather_mean_kernel(const int* __restrict__ nodes,
                   const int* __restrict__ neigh_idx,
                   const float* __restrict__ table_self,
                   const float* __restrict__ tables_to,
                   unsigned short* __restrict__ X) {
    const int n = blockIdx.x;
    const int t = threadIdx.x;
    const int s = t >> 6;     // 0..2: relation, 3: self
    const int q = t & 63;     // float4 index within D

    float4 acc = make_float4(0.f, 0.f, 0.f, 0.f);

    if (s < R) {
        const int* __restrict__ idx = neigh_idx + ((size_t)s * N + (size_t)n) * K;
        const float4* __restrict__ T =
            (const float4*)tables_to + (size_t)s * NTOT * (D / 4);
        int row = idx[0];
        #pragma unroll 5
        for (int k = 0; k < K; ++k) {
            const int nrow = (k + 1 < K) ? idx[k + 1] : row;
            // gfx1250 global_prefetch_b8: pull next gathered row toward L2
            __builtin_prefetch(&T[(size_t)nrow * 64 + q], 0, 0);
            float4 v = T[(size_t)row * 64 + q];
            acc.x += v.x; acc.y += v.y; acc.z += v.z; acc.w += v.w;
            row = nrow;
        }
        const float inv = 1.0f / (float)K;
        acc.x *= inv; acc.y *= inv; acc.z *= inv; acc.w *= inv;
    } else {
        const int row = nodes[n];
        acc = ((const float4*)table_self)[(size_t)row * 64 + q];
    }

    const unsigned int lo = pack2bf(acc.x, acc.y);
    const unsigned int hi = pack2bf(acc.z, acc.w);
    const size_t e = (size_t)n * KC + (size_t)s * D + (size_t)q * 4;
    *(uint2*)(X + e) = make_uint2(lo, hi);
}

// ---------------------------------------------------------------------------
// Kernel 2: fuse weights.  Mcat[o, s*256 + d] = sum_j W_comp[o, s*256+j] *
// Wr_s[j, d]   (Wr_3 = W_self).  Then h = relu(X @ Mcat^T) reproduces the
// rel-matmul + concat + compress exactly (halves GEMM FLOPs to 8.6 G).
// ---------------------------------------------------------------------------
__global__ void __launch_bounds__(128)
fuse_weights_kernel(const float* __restrict__ W_self,
                    const float* __restrict__ W_rel,
                    const float* __restrict__ W_comp,
                    unsigned short* __restrict__ Mcat) {
    const int b = blockIdx.x;          // 1024 blocks
    const int o = b >> 2;              // output row 0..255
    const int s = b & 3;               // section 0..3
    const int d0 = threadIdx.x * 2;    // two d per thread

    const float* __restrict__ Wr =
        (s < R) ? (W_rel + (size_t)s * OUT * D) : W_self;
    const float* __restrict__ wc = W_comp + (size_t)o * KC + (size_t)s * D;

    float a0 = 0.f, a1 = 0.f;
    #pragma unroll 4
    for (int j = 0; j < OUT; ++j) {
        const float  c = wc[j];                               // uniform -> s_load
        const float2 w = *(const float2*)(Wr + (size_t)j * D + d0);
        a0 = fmaf(c, w.x, a0);
        a1 = fmaf(c, w.y, a1);
    }
    *(unsigned int*)(Mcat + (size_t)o * KC + (size_t)s * D + d0) = pack2bf(a0, a1);
}

// ---------------------------------------------------------------------------
// Kernel 3: h = relu(X @ Mcat^T) fused with the 8-row bag-mean.
//
// Block tile 64(M) x 128(N), 8 waves, each wave a 32x32 tile = 2x2 WMMA
// subtiles (4 accumulators) -> 4 independent v_wmma_f32_16x16x32_bf16 per
// k-chunk from 2 A-frags + 2 B-frags (halves loads-per-WMMA vs 1-tile waves).
//
// A (X rows, 32 MB streamed once) is staged global->LDS with async b128
// copies, double-buffered over k-chunks of 32: chunk = 64 rows x 64B = 4KB,
// one b128 per thread.  Removes the 4x redundant A reads across waves and
// bypasses VGPRs on the copy.  B (Mcat, 512 KB, L2-resident) loads straight
// from global.
//
// Fragment layouts (per ISA 7.12.2), pure lane-addressed b128s:
//   A 16x32 bf16: lane l<16 = row l, K {0..7,16..23}; lane l+16 = row l,
//                 K {8..15,24..31} -> ds_load_b128 at sel*16 and sel*16+32.
//   B 32x16 bf16: lane l<16 = col l, K {0..7,16..23}; lane l+16 = col l,
//                 K {8..15,24..31} -> two global b128 from Mcat row.
// Epilogue: tile rows 0..7 sit in c[0..7] of lanes 0..15, rows 8..15 in
// lanes 16..31 -> per-lane relu+sum is exactly one bag, one store per lane.
// ---------------------------------------------------------------------------
constexpr int BLK_M = 64;
constexpr int BLK_N = 128;
constexpr int KCH   = 32;                 // k elements per staged chunk
constexpr int CHB   = BLK_M * KCH * 2;    // 4096 bytes per LDS buffer

__global__ void __launch_bounds__(256)
wmma_gemm_bag_kernel(const __bf16* __restrict__ X,
                     const __bf16* __restrict__ Mcat,
                     float* __restrict__ out) {
    __shared__ __align__(16) unsigned char ldsA[2][CHB];   // 8 KB double buffer

    const int t    = threadIdx.x;
    const int lane = t & 31;
    const int wave = t >> 5;
    const int wm   = wave >> 2;          // 0..1
    const int wn   = wave & 3;           // 0..3
    const int l15  = lane & 15;
    const int sel  = lane >> 4;

    const int row_blk = blockIdx.x * BLK_M;
    const int col0    = blockIdx.y * BLK_N + wn * 32;

    // producer mapping: thread t copies 16B: row (t>>2), k-segment (t&3)
    const int grow = t >> 2;
    const int gseg = t & 3;
    const __bf16* gsrc = X + (size_t)(row_blk + grow) * KC + gseg * 8;
    const int lofs = grow * 64 + gseg * 16;

    // A consumer LDS byte offsets (row-in-block * 64 + sel*16)
    const int a_off0 = (wm * 32 + l15) * 64 + sel * 16;
    const int a_off1 = a_off0 + 16 * 64;

    // B global pointers for the two N subtiles
    const __bf16* bp0 = Mcat + (size_t)(col0 + l15) * KC + sel * 8;
    const __bf16* bp1 = bp0 + (size_t)16 * KC;

    v8f c00 = {}, c01 = {}, c10 = {}, c11 = {};

    // prologue: stage chunk 0
    async_copy16(gsrc, &ldsA[0][lofs]);
    async_wait_all();
    __syncthreads();

    for (int kc = 0; kc < KC / KCH; ++kc) {          // 32 chunks
        const int cur = kc & 1;
        if (kc + 1 < KC / KCH)
            async_copy16(gsrc + (size_t)(kc + 1) * KCH, &ldsA[cur ^ 1][lofs]);

        const unsigned char* La = &ldsA[cur][0];
        const v16bf A0 = cat8(*(const v8bf*)(La + a_off0),
                              *(const v8bf*)(La + a_off0 + 32));
        const v16bf A1 = cat8(*(const v8bf*)(La + a_off1),
                              *(const v8bf*)(La + a_off1 + 32));

        const int kb = kc * KCH;
        const v16bf B0 = cat8(*(const v8bf*)(bp0 + kb),
                              *(const v8bf*)(bp0 + kb + 16));
        const v16bf B1 = cat8(*(const v8bf*)(bp1 + kb),
                              *(const v8bf*)(bp1 + kb + 16));

        c00 = __builtin_amdgcn_wmma_f32_16x16x32_bf16(false, A0, false, B0,
                                                      (short)0, c00, false, false);
        c01 = __builtin_amdgcn_wmma_f32_16x16x32_bf16(false, A0, false, B1,
                                                      (short)0, c01, false, false);
        c10 = __builtin_amdgcn_wmma_f32_16x16x32_bf16(false, A1, false, B0,
                                                      (short)0, c10, false, false);
        c11 = __builtin_amdgcn_wmma_f32_16x16x32_bf16(false, A1, false, B1,
                                                      (short)0, c11, false, false);

        async_wait_all();
        __syncthreads();
    }

    // relu + bag-of-8 mean straight out of the C-tile register layout
    float s00 = 0.f, s01 = 0.f, s10 = 0.f, s11 = 0.f;
    #pragma unroll
    for (int i = 0; i < 8; ++i) {
        s00 += fmaxf(c00[i], 0.f);
        s01 += fmaxf(c01[i], 0.f);
        s10 += fmaxf(c10[i], 0.f);
        s11 += fmaxf(c11[i], 0.f);
    }
    const float inv = 1.0f / (float)BAG;
    const int m0   = row_blk + wm * 32;          // rows of subtile mi=0
    const int or0  = (m0 >> 3) + sel;            // bag row for mi=0
    const int or1  = or0 + 2;                    // +16 rows = +2 bags
    const int oc0  = col0 + l15;
    const int oc1  = oc0 + 16;
    out[(size_t)or0 * OUT + oc0] = s00 * inv;
    out[(size_t)or0 * OUT + oc1] = s01 * inv;
    out[(size_t)or1 * OUT + oc0] = s10 * inv;
    out[(size_t)or1 * OUT + oc1] = s11 * inv;
}

// ---------------------------------------------------------------------------
// Launch: K1 (gather) and K2 (weight fuse) independent; K3 consumes both.
// Workspace: X bf16 [16384,1024] = 32 MiB, then Mcat bf16 [256,1024] = 512 KiB.
// ---------------------------------------------------------------------------
extern "C" void kernel_launch(void* const* d_in, const int* in_sizes, int n_in,
                              void* d_out, int out_size, void* d_ws, size_t ws_size,
                              hipStream_t stream) {
    const int*   nodes      = (const int*)  d_in[0];
    const int*   neigh_idx  = (const int*)  d_in[1];
    const float* table_self = (const float*)d_in[2];
    const float* tables_to  = (const float*)d_in[3];
    const float* W_self     = (const float*)d_in[4];
    const float* W_rel      = (const float*)d_in[5];
    const float* W_comp     = (const float*)d_in[6];
    float* out = (float*)d_out;

    unsigned short* X    = (unsigned short*)d_ws;
    unsigned short* Mcat = X + (size_t)N * KC;     // +32 MiB

    gather_mean_kernel<<<N, 256, 0, stream>>>(nodes, neigh_idx, table_self,
                                              tables_to, X);
    fuse_weights_kernel<<<OUT * 4, 128, 0, stream>>>(W_self, W_rel, W_comp, Mcat);

    dim3 grid(N / BLK_M, OUT / BLK_N);   // 256 x 2 blocks, 8 waves each
    wmma_gemm_bag_kernel<<<grid, 256, 0, stream>>>((const __bf16*)X,
                                                   (const __bf16*)Mcat, out);
}
